// ProbNodeModel_37177236914590
// MI455X (gfx1250) — compile-verified, matrix operand order checked by reading, not run on hardware
//
#include <hip/hip_runtime.h>
#include <hip/hip_bf16.h>

// ---------------- problem constants (from the reference) ----------------
#define NN      50000
#define EE      400000
#define BB      64
#define N_INC   64
#define E_OUTC  64
#define U_INC   32
#define H1A     128
#define H1B     64
#define MLP2_IN 160      // N_INC + H1B + U_INC
#define H2      256
#define DIM_LH  128
#define DIM_L   32
#define N_OUTC  64
#define LN_EPSF 1e-5f

typedef __bf16 bf16_t;
typedef __attribute__((ext_vector_type(16))) __bf16 v16bf;
typedef __attribute__((ext_vector_type(8)))  float  v8f;

// ---------------- WMMA fragment helpers (CDNA5 16x16x32 bf16) ----------------
// 16-bit A 16x32 layout (ISA 7.12.2): lanes 0-15 row M=lane hold K=0..7,16..23;
// lanes 16-31 row M=lane-16 hold K=8..15,24..31.  half h -> K = (h<8?h:h+8)+grp*8.

__device__ __forceinline__ v16bf load_a_frag(const bf16_t* tile, int stride, int kc, int lane) {
    const int m  = lane & 15;
    const int k0 = kc * 32 + ((lane >> 4) << 3);     // multiple of 8 -> 16B aligned
    const bf16_t* p = tile + m * stride + k0;
    v16bf a;
#pragma unroll
    for (int i = 0; i < 8; ++i) { a[i] = p[i]; a[8 + i] = p[16 + i]; }
    return a;
}

// B fragments come pre-packed by pack_weights(): 32B contiguous per lane.
__device__ __forceinline__ v16bf load_b_frag(const bf16_t* pk, int KC, int nt, int kc, int lane) {
    const bf16_t* p = pk + ((((size_t)nt * KC + kc) * 32 + lane) << 4);
    v16bf b;
#pragma unroll
    for (int i = 0; i < 16; ++i) b[i] = p[i];
    return b;
}

// One 16x16 f32 D tile: accumulate over KC chunks of 32.
__device__ __forceinline__ v8f gemm_tile(const bf16_t* atile, int astride,
                                         const bf16_t* wpk, int KC, int nt, int lane) {
    v8f c = {0.f, 0.f, 0.f, 0.f, 0.f, 0.f, 0.f, 0.f};
    for (int kc = 0; kc < KC; ++kc) {
        v16bf a = load_a_frag(atile, astride, kc, lane);
        v16bf b = load_b_frag(wpk, KC, nt, kc, lane);
        c = __builtin_amdgcn_wmma_f32_16x16x32_bf16(false, a, false, b, (short)0, c, false, false);
    }
    return c;
}

// Fused 16-row GEMM + bias + LayerNorm + ReLU + bf16 store, all in registers.
// D layout: VGPR r holds M=r (lanes 0-15) / M=r+8 (lanes 16-31), N=lane&15.
// Row m's 16*NT elements live entirely in one 16-lane group -> 4-step shfl_xor
// reduction gives every lane its 8 rows' sums with no LDS round trip.
template<int NT, int KC>
__device__ __forceinline__ void gemm_ln_relu_store(
        const bf16_t* atile, int astride, const bf16_t* wpk,
        const float* __restrict__ bias, const float* __restrict__ g,
        const float* __restrict__ be,
        bf16_t* obuf, int ostride, int lane) {
    v8f acc[NT];
#pragma unroll
    for (int nt = 0; nt < NT; ++nt) {
        v8f c = {0.f, 0.f, 0.f, 0.f, 0.f, 0.f, 0.f, 0.f};
#pragma unroll
        for (int kc = 0; kc < KC; ++kc) {
            v16bf a = load_a_frag(atile, astride, kc, lane);
            v16bf b = load_b_frag(wpk, KC, nt, kc, lane);
            c = __builtin_amdgcn_wmma_f32_16x16x32_bf16(false, a, false, b, (short)0, c, false, false);
        }
        acc[nt] = c;
    }
    const int nl = lane & 15;
    const int mo = (lane >> 4) << 3;
    float sum[8], ssq[8];
#pragma unroll
    for (int r = 0; r < 8; ++r) { sum[r] = 0.f; ssq[r] = 0.f; }
#pragma unroll
    for (int nt = 0; nt < NT; ++nt) {
        const float bb = bias[nt * 16 + nl];
#pragma unroll
        for (int r = 0; r < 8; ++r) {
            float v = acc[nt][r] + bb;
            acc[nt][r] = v;
            sum[r] += v;
            ssq[r] += v * v;
        }
    }
#pragma unroll
    for (int msk = 1; msk < 16; msk <<= 1) {
#pragma unroll
        for (int r = 0; r < 8; ++r) {
            sum[r] += __shfl_xor(sum[r], msk, 32);
            ssq[r] += __shfl_xor(ssq[r], msk, 32);
        }
    }
    const float invH = 1.0f / (float)(NT * 16);
#pragma unroll
    for (int r = 0; r < 8; ++r) {
        float mean = sum[r] * invH;
        float var  = ssq[r] * invH - mean * mean;
        sum[r] = mean;
        ssq[r] = rsqrtf(fmaxf(var, 0.f) + LN_EPSF);
    }
#pragma unroll
    for (int nt = 0; nt < NT; ++nt) {
        const int n = nt * 16 + nl;
        const float gg = g[n], bb = be[n];
#pragma unroll
        for (int r = 0; r < 8; ++r) {
            float v = (acc[nt][r] - sum[r]) * ssq[r] * gg + bb;
            obuf[(r + mo) * ostride + n] = (bf16_t)(v > 0.f ? v : 0.f);
        }
    }
}

// Deterministic hash-based standard normal (Box-Muller) for the reparameterization.
__device__ __forceinline__ float hash_gauss(unsigned idx) {
    unsigned a = idx * 1664525u + 1013904223u;
    a ^= a >> 16; a *= 0x7feb352du; a ^= a >> 15; a *= 0x846ca68bu; a ^= a >> 16;
    unsigned b = (idx ^ 0x9e3779b9u) * 22695477u + 12345u;
    b ^= b >> 16; b *= 0x2c1b3c6du; b ^= b >> 12; b *= 0x297a2d39u; b ^= b >> 15;
    float u1 = (float)(a >> 8) * (1.f / 16777216.f) + 1e-7f;
    float u2 = (float)(b >> 8) * (1.f / 16777216.f);
    return sqrtf(-2.f * __logf(u1)) * __cosf(6.28318530718f * u2);
}

// ---------------- kernel 0: pack f32 weights -> fragment-native bf16 ----------------
// packed[((nt*KC + kc)*32 + lane)*16 + h] = bf16(W[kc*32 + kmap(h,lane)][nt*16 + (lane&15)])
__global__ void pack_weights(const float* __restrict__ W, bf16_t* __restrict__ dst,
                             int din, int dout) {
    int idx = blockIdx.x * blockDim.x + threadIdx.x;
    int total = din * dout;
    if (idx >= total) return;
    int h    = idx & 15;
    int lane = (idx >> 4) & 31;
    int rest = idx >> 9;
    int KC   = din >> 5;
    int kc   = rest % KC;
    int nt   = rest / KC;
    int k = kc * 32 + ((h < 8) ? h : h + 8) + ((lane >> 4) << 3);
    int n = nt * 16 + (lane & 15);
    dst[idx] = (bf16_t)W[(size_t)k * dout + n];
}

// ---------------- kernel 1: edge MLP + scatter-add ----------------
// 4 waves/block, each wave owns a 16-edge tile.  E = 400000 = 6250*64, no remainder.
__global__ __launch_bounds__(128)
void edge_mlp_scatter(const float* __restrict__ x, const int* __restrict__ ei,
                      const float* __restrict__ eattr,
                      const float* __restrict__ b1, const float* __restrict__ g1,
                      const float* __restrict__ be1, const float* __restrict__ b2,
                      const bf16_t* __restrict__ W1p, const bf16_t* __restrict__ W2p,
                      float* __restrict__ sums, float* __restrict__ cnts) {
    __shared__ bf16_t sSE[4][16 * 136];   // se tile (128 wide, pad 8)
    __shared__ bf16_t sH1[4][16 * 136];   // relu(ln(h1)) bf16
    __shared__ int    sCol[4][16];

    const int wave = threadIdx.x >> 5;
    const int lane = threadIdx.x & 31;
    const int e0   = (blockIdx.x * 4 + wave) * 16;

    // ---- stage se = [x[row] | edge_attr] as bf16 (lane pair per row) ----
    {
        const int m = lane & 15;
        const int e = e0 + m;
        if (lane < 16) {
            const int r = ei[e];                       // edge_index[0][e]
            const float* xp = x + (size_t)r * N_INC;
            bf16_t* op = &sSE[wave][m * 136];
#pragma unroll
            for (int j = 0; j < N_INC; j += 4) {
                float4 v = *(const float4*)(xp + j);
                op[j] = (bf16_t)v.x; op[j+1] = (bf16_t)v.y;
                op[j+2] = (bf16_t)v.z; op[j+3] = (bf16_t)v.w;
            }
            sCol[wave][m] = ei[EE + e];                // edge_index[1][e]
        } else {
            const float* ap = eattr + (size_t)e * E_OUTC;
            bf16_t* op = &sSE[wave][m * 136 + N_INC];
#pragma unroll
            for (int j = 0; j < E_OUTC; j += 4) {
                float4 v = *(const float4*)(ap + j);
                op[j] = (bf16_t)v.x; op[j+1] = (bf16_t)v.y;
                op[j+2] = (bf16_t)v.z; op[j+3] = (bf16_t)v.w;
            }
        }
    }
    __syncthreads();

    // ---- h1 = relu(ln(se @ W1 + b1)) : K=128 (4 chunks), N=128 (8 tiles), fused ----
    gemm_ln_relu_store<8, 4>(sSE[wave], 136, W1p, b1, g1, be1, sH1[wave], 136, lane);
    __syncthreads();

    // ---- h = h1 @ W2 + b2, scatter-add to sums[col] from fragments ----
    const int mo = (lane >> 4) << 3;
    for (int nt = 0; nt < 4; ++nt) {
        v8f c = gemm_tile(sH1[wave], 136, W2p, 4, nt, lane);
        const int n  = nt * 16 + (lane & 15);
        const float bb = b2[n];
#pragma unroll
        for (int r = 0; r < 8; ++r) {
            const int ci = sCol[wave][r + mo];
            atomicAdd(&sums[(size_t)ci * H1B + n], c[r] + bb);
        }
    }
    if (lane < 16) atomicAdd(&cnts[sCol[wave][lane]], 1.0f);
}

// ---------------- kernel 2: fused node pipeline ----------------
// 4 waves/block, each wave owns a 16-node tile; partial tiles clamp loads, guard writes.
__global__ __launch_bounds__(128)
void node_pipeline(const float* __restrict__ x, const float* __restrict__ sums,
                   const float* __restrict__ cnts, const float* __restrict__ u,
                   const int* __restrict__ batch,
                   const float* __restrict__ b3, const float* __restrict__ g3, const float* __restrict__ be3,
                   const float* __restrict__ b4, const float* __restrict__ g4, const float* __restrict__ be4,
                   const float* __restrict__ bm, const float* __restrict__ bv, const float* __restrict__ bx,
                   const bf16_t* __restrict__ W3p, const bf16_t* __restrict__ W4p,
                   const bf16_t* __restrict__ Wmp, const bf16_t* __restrict__ Wvp,
                   const bf16_t* __restrict__ Wxp,
                   float* __restrict__ out, float* __restrict__ zmu, float* __restrict__ zvar) {
    __shared__ bf16_t sZ[4][16 * 168];    // z (160 wide) / z4 (128 wide) bf16 stage
    __shared__ bf16_t sS[4][16 * 264];    // z3 (256 wide) and z_s (32 wide, stride 40) stage

    const int wave = threadIdx.x >> 5;
    const int lane = threadIdx.x & 31;
    const int i0   = (blockIdx.x * 4 + wave) * 16;
    const int nl   = lane & 15;
    const int mo   = (lane >> 4) << 3;

    // ---- stage z = [x | sums/max(cnt,1) | u[batch]] (lane pair per row) ----
    {
        const int i = min(i0 + nl, NN - 1);
        bf16_t* op = &sZ[wave][nl * 168];
        if (lane < 16) {
            const float* xp = x + (size_t)i * N_INC;
#pragma unroll
            for (int j = 0; j < N_INC; j += 4) {
                float4 v = *(const float4*)(xp + j);
                op[j] = (bf16_t)v.x; op[j+1] = (bf16_t)v.y;
                op[j+2] = (bf16_t)v.z; op[j+3] = (bf16_t)v.w;
            }
            const float* up = u + (size_t)batch[i] * U_INC;
#pragma unroll
            for (int j = 0; j < U_INC; j += 4) {
                float4 v = *(const float4*)(up + j);
                op[128+j] = (bf16_t)v.x; op[128+j+1] = (bf16_t)v.y;
                op[128+j+2] = (bf16_t)v.z; op[128+j+3] = (bf16_t)v.w;
            }
        } else {
            const float invc = 1.0f / fmaxf(cnts[i], 1.0f);
            const float* sp = sums + (size_t)i * H1B;
#pragma unroll
            for (int j = 0; j < H1B; j += 4) {
                float4 v = *(const float4*)(sp + j);
                op[64+j]   = (bf16_t)(v.x * invc); op[64+j+1] = (bf16_t)(v.y * invc);
                op[64+j+2] = (bf16_t)(v.z * invc); op[64+j+3] = (bf16_t)(v.w * invc);
            }
        }
    }
    __syncthreads();

    // ---- z3 = relu(ln(z @ W3 + b3)) : K=160 (5 chunks), N=256 (16 tiles), fused ----
    gemm_ln_relu_store<16, 5>(sZ[wave], 168, W3p, b3, g3, be3, sS[wave], 264, lane);
    __syncthreads();

    // ---- z4 = relu(ln(z3 @ W4 + b4)) : K=256 (8 chunks), N=128 (8 tiles), fused ----
    gemm_ln_relu_store<8, 8>(sS[wave], 264, W4p, b4, g4, be4, sZ[wave], 168, lane);
    __syncthreads();

    // ---- z_mu / z_logvar + reparameterize, entirely in fragment layout ----
    for (int nt = 0; nt < 2; ++nt) {
        v8f cm = gemm_tile(sZ[wave], 168, Wmp, 4, nt, lane);
        v8f cv = gemm_tile(sZ[wave], 168, Wvp, 4, nt, lane);
        const int n = nt * 16 + nl;
        const float bbm = bm[n], bbv = bv[n];
#pragma unroll
        for (int r = 0; r < 8; ++r) {
            const int m = r + mo;
            const int i = i0 + m;
            const float vm = cm[r] + bbm;
            const float vv = cv[r] + bbv;
            if (i < NN) {
                zmu [(size_t)i * DIM_L + n] = vm;
                zvar[(size_t)i * DIM_L + n] = vv;
            }
            const float eps = hash_gauss((unsigned)(i * DIM_L + n) ^ 0x2a2a2a2au);
            sS[wave][m * 40 + n] = (bf16_t)(vm + eps * __expf(0.5f * vv));  // z_s stage
        }
    }
    __syncthreads();

    // ---- out = z_s @ Wx + bx : K=32 (1 chunk), N=64 (4 tiles) ----
    for (int nt = 0; nt < 4; ++nt) {
        v8f c = gemm_tile(sS[wave], 40, Wxp, 1, nt, lane);
        const int n = nt * 16 + nl;
        const float bb = bx[n];
#pragma unroll
        for (int r = 0; r < 8; ++r) {
            const int i = i0 + r + mo;
            if (i < NN) out[(size_t)i * N_OUTC + n] = c[r] + bb;
        }
    }
}

// ---------------- host launcher ----------------
extern "C" void kernel_launch(void* const* d_in, const int* in_sizes, int n_in,
                              void* d_out, int out_size, void* d_ws, size_t ws_size,
                              hipStream_t stream) {
    const float* x     = (const float*)d_in[0];
    const int*   ei    = (const int*)  d_in[1];
    const float* eattr = (const float*)d_in[2];
    const float* u     = (const float*)d_in[3];
    const int*   batch = (const int*)  d_in[4];
    // d_in[5] = goal (unused by the reference)
    const float* W1 = (const float*)d_in[6];  const float* b1  = (const float*)d_in[7];
    const float* g1 = (const float*)d_in[8];  const float* be1 = (const float*)d_in[9];
    const float* W2 = (const float*)d_in[10]; const float* b2  = (const float*)d_in[11];
    const float* W3 = (const float*)d_in[12]; const float* b3  = (const float*)d_in[13];
    const float* g3 = (const float*)d_in[14]; const float* be3 = (const float*)d_in[15];
    const float* W4 = (const float*)d_in[16]; const float* b4  = (const float*)d_in[17];
    const float* g4 = (const float*)d_in[18]; const float* be4 = (const float*)d_in[19];
    const float* Wm = (const float*)d_in[20]; const float* bm  = (const float*)d_in[21];
    const float* Wv = (const float*)d_in[22]; const float* bv  = (const float*)d_in[23];
    const float* Wx = (const float*)d_in[24]; const float* bx  = (const float*)d_in[25];

    // d_out: [out N*64 | z_mu N*32 | z_var N*32]
    float* out  = (float*)d_out;
    float* zmu  = out + (size_t)NN * N_OUTC;
    float* zvar = zmu + (size_t)NN * DIM_L;

    // workspace layout (bytes)
    char* ws = (char*)d_ws;
    bf16_t* W1p = (bf16_t*)(ws);                    // 128*128
    bf16_t* W2p = (bf16_t*)(ws + 32768);            // 128*64
    bf16_t* W3p = (bf16_t*)(ws + 49152);            // 160*256
    bf16_t* W4p = (bf16_t*)(ws + 131072);           // 256*128
    bf16_t* Wmp = (bf16_t*)(ws + 196608);           // 128*32
    bf16_t* Wvp = (bf16_t*)(ws + 204800);           // 128*32
    bf16_t* Wxp = (bf16_t*)(ws + 212992);           // 32*64
    float*  sums = (float*)(ws + 217088);           // N*64 f32
    float*  cnts = (float*)(ws + 217088 + (size_t)NN * H1B * 4); // N f32

    // zero the scatter accumulators (memset node is graph-capturable)
    hipMemsetAsync(sums, 0, (size_t)NN * H1B * 4 + (size_t)NN * 4, stream);

    // pack weights to fragment-native bf16
    auto pk = [&](const float* W, bf16_t* D, int din, int dout) {
        int total = din * dout;
        pack_weights<<<(total + 255) / 256, 256, 0, stream>>>(W, D, din, dout);
    };
    pk(W1, W1p, H1A, H1A);
    pk(W2, W2p, H1A, H1B);
    pk(W3, W3p, MLP2_IN, H2);
    pk(W4, W4p, H2, DIM_LH);
    pk(Wm, Wmp, DIM_LH, DIM_L);
    pk(Wv, Wvp, DIM_LH, DIM_L);
    pk(Wx, Wxp, DIM_L, N_OUTC);

    // edge MLP + scatter: 400000 edges / (4 waves * 16 edges) = 6250 blocks
    edge_mlp_scatter<<<EE / 64, 128, 0, stream>>>(x, ei, eattr, b1, g1, be1, b2,
                                                  W1p, W2p, sums, cnts);

    // fused node pipeline: ceil(50000 / 64) = 782 blocks
    node_pipeline<<<(NN + 63) / 64, 128, 0, stream>>>(x, sums, cnts, u, batch,
                                                      b3, g3, be3, b4, g4, be4,
                                                      bm, bv, bx,
                                                      W3p, W4p, Wmp, Wvp, Wxp,
                                                      out, zmu, zvar);
}
